// Mul_HeadSA_62173946577326
// MI455X (gfx1250) — matrix operand order repaired and checked by reading, not compiled, not run on hardware
//
#include <hip/hip_runtime.h>
#include <hip/hip_bf16.h>

typedef _Float16 f16;
typedef __attribute__((ext_vector_type(16))) _Float16 v16h;
typedef __attribute__((ext_vector_type(8)))  _Float16 v8h;
typedef __attribute__((ext_vector_type(8)))  float    v8f;
typedef __attribute__((ext_vector_type(4)))  float    v4f;

#define DIM   1024
#define NH    16
#define HD2   64
#define TT    2048
#define SSZ   2048
#define BB    2
#define MROWS 4096   // T*B == S*B

// ---------------- WMMA fragment helpers (ISA 7.12.2 layouts) ----------------
// Global loads: uniform base + 32-bit BYTE offset -> saddr/voffset form.
// A-frag 16x32 f16: lane row M=lane%16; K = g*8+[0..7] and 16+g*8+[0..7]
//   -> bytes g*16 and 32+g*16 off the row base.
__device__ __forceinline__ v16h load_a_g(const f16* __restrict__ base,
                                         unsigned byteoff, int g) {
  const char* p = (const char*)base + byteoff;
  v8h lo = *(const v8h*)(p + g * 16);
  v8h hi = *(const v8h*)(p + 32 + g * 16);
  v16h r;
#pragma unroll
  for (int j = 0; j < 8; ++j) { r[j] = lo[j]; r[8 + j] = hi[j]; }
  return r;
}
// B-frag 32x16 f16: lane column N=lane%16; K = g*8+[0..7] and 16+g*8+[0..7]
//   -> bytes g*16 and 32+g*16 off the column base.
__device__ __forceinline__ v16h load_b_g(const f16* __restrict__ base,
                                         unsigned byteoff, int g) {
  const char* p = (const char*)base + byteoff;
  v8h lo = *(const v8h*)(p + g * 16);
  v8h hi = *(const v8h*)(p + 32 + g * 16);
  v16h r;
#pragma unroll
  for (int j = 0; j < 8; ++j) { r[j] = lo[j]; r[8 + j] = hi[j]; }
  return r;
}
// A-frag from LDS (f16 pointer into shared)
__device__ __forceinline__ v16h load_a_lds(const f16* p, int g) {
  v8h lo = *(const v8h*)(p + g * 8);
  v8h hi = *(const v8h*)(p + 16 + g * 8);
  v16h r;
#pragma unroll
  for (int j = 0; j < 8; ++j) { r[j] = lo[j]; r[8 + j] = hi[j]; }
  return r;
}
__device__ __forceinline__ v8f wmma16(v16h a, v16h b, v8f c) {
  return __builtin_amdgcn_wmma_f32_16x16x32_f16(false, a, false, b, (short)0, c,
                                                false, false);
}

// ---------------- f32 -> f16 pre-conversion (bandwidth-trivial) -------------
__global__ void cvt_kernel(const float* __restrict__ in, f16* __restrict__ out,
                           int n) {
  const int i = (blockIdx.x * blockDim.x + threadIdx.x) * 8;
  if (i + 8 <= n) {
    v4f a = *(const v4f*)(in + i);
    v4f b = *(const v4f*)(in + i + 4);
    v8h o;
#pragma unroll
    for (int j = 0; j < 4; ++j) { o[j] = (f16)a[j]; o[4 + j] = (f16)b[j]; }
    *(v8h*)(out + i) = o;
  }
}

// ---------------- Projection / output GEMM: C = A @ W^T + bias --------------
// Block 128 = 4 waves; wave tile 32x64 (2 A-frags x 4 B-frags, 8 WMMA/k-step).
// MODE 0: Q proj  -> f16 [M,DIM], (acc+bias)*scale
// MODE 1: K proj  -> f16 [M,DIM]
// MODE 2: V proj  -> f16 transposed Vt[B,NH,HD,S]
// MODE 3: out proj -> f32 [M,DIM] (d_out)
template <int MODE>
__global__ __launch_bounds__(128)
void gemm_kernel(const f16* __restrict__ A, const f16* __restrict__ W,
                 const float* __restrict__ bias, void* __restrict__ Cptr,
                 float scale) {
  const int lane = threadIdx.x & 31;
  const int wave = threadIdx.x >> 5;
  const int g    = lane >> 4;
  const int ln   = lane & 15;
  const int m0   = blockIdx.x * 128 + wave * 32;
  const int n0   = blockIdx.y * 64;

  const unsigned aoff0 = (unsigned)(m0 + ln) * (DIM * 2);       // byte offsets
  const unsigned aoff1 = aoff0 + 16u * DIM * 2;
  unsigned woff[4];
#pragma unroll
  for (int nt = 0; nt < 4; ++nt)
    woff[nt] = (unsigned)(n0 + nt * 16 + ln) * (DIM * 2);

  v8f acc[2][4] = {};
  for (int k0 = 0; k0 < DIM * 2; k0 += 64) {                    // bytes
    const v16h a0 = load_a_g(A, aoff0 + k0, g);
    const v16h a1 = load_a_g(A, aoff1 + k0, g);
#pragma unroll
    for (int nt = 0; nt < 4; ++nt) {
      const v16h bf = load_b_g(W, woff[nt] + k0, g);
      acc[0][nt] = wmma16(a0, bf, acc[0][nt]);
      acc[1][nt] = wmma16(a1, bf, acc[1][nt]);
    }
  }

#pragma unroll
  for (int mt = 0; mt < 2; ++mt) {
#pragma unroll
    for (int nt = 0; nt < 4; ++nt) {
      const int n  = n0 + nt * 16 + ln;
      const float bv = bias[n];
#pragma unroll
      for (int r = 0; r < 8; ++r) {
        const int m = m0 + mt * 16 + r + g * 8;
        const float v = (acc[mt][nt][r] + bv) * scale;
        if constexpr (MODE == 0 || MODE == 1) {
          ((f16*)Cptr)[(unsigned)m * DIM + n] = (f16)v;
        } else if constexpr (MODE == 2) {
          const int s = m >> 1, b = m & 1;          // rows are s*B + b
          const int h = n >> 6, d = n & 63;
          ((f16*)Cptr)[((unsigned)(b * NH + h) * HD2 + d) * SSZ + s] = (f16)v;
        } else {
          ((float*)Cptr)[(unsigned)m * DIM + n] = v;
        }
      }
    }
  }
}

// ---------------- Fused attention ------------------------------------------
// grid (T/16, B), block 128 (4 waves). Dynamic LDS (pE FIRST so its byte
// addresses fit the 16-bit DS immediate-offset field; 8 row stores share one
// address register):
//   pE f16[16][SSZ] (bytes 0..65535) | avg f32[16][SSZ] | red/rowmax/rowinv
// pE holds raw scores then unnormalized exp; PV normalization folded into the
// output store; head-average accumulated in LDS, written to global once.
__global__ __launch_bounds__(128)
void attn_kernel(const f16* __restrict__ Qh, const f16* __restrict__ Kh,
                 const f16* __restrict__ Vt, f16* __restrict__ Oh,
                 float* __restrict__ attn_avg) {
  extern __shared__ float smem[];
  f16*   pE     = (f16*)smem;                    // 16*SSZ f16  (LDS offset 0)
  float* avg    = smem + 8 * SSZ;                // 16*SSZ f32
  float* red    = smem + 24 * SSZ;               // 128
  float* rowmax = red + 128;                     // 16
  float* rowinv = rowmax + 16;                   // 16

  const int tid  = threadIdx.x;
  const int lane = tid & 31;
  const int wave = tid >> 5;
  const int g    = lane >> 4;
  const int ln   = lane & 15;
  const int b    = blockIdx.y;
  const int t0   = blockIdx.x * 16;
  const int r3   = tid >> 3;                     // owned row 0..15
  const int c3   = (tid & 7) * 256;              // owned 256-col span
  const float invH = 1.0f / NH;

  for (int h = 0; h < NH; ++h) {
    // ---- Phase 1: raw scores (f16) = Q_tile(16x64) @ K^T, 2 WMMA k-steps
    const unsigned qoff =
        (((unsigned)(t0 + ln) * BB + b) * DIM + h * HD2) * 2;   // bytes
    const v16h aq0 = load_a_g(Qh, qoff, g);
    const v16h aq1 = load_a_g(Qh, qoff + 64, g);
    unsigned koff =
        (((unsigned)(wave * 512 + ln) * BB + b) * DIM + h * HD2) * 2;
    for (int i = 0; i < 32; ++i) {               // wave owns 512 s-cols
      const int s0 = wave * 512 + i * 16;
      v8f c = {};
      c = wmma16(aq0, load_b_g(Kh, koff, g), c);
      c = wmma16(aq1, load_b_g(Kh, koff + 64, g), c);
      koff += 16u * BB * DIM * 2;                // 64 KiB per 16 s-rows
      f16* st = pE + (unsigned)(g * 8) * SSZ + s0 + ln;
#pragma unroll
      for (int r = 0; r < 8; ++r)
        st[r * SSZ] = (f16)c[r];                 // ds_store offset:r*4096
    }
    __syncthreads();

    // ---- Phase 2: row max, exp (in place, unnormalized), 1/sum
    {
      const f16* pr = pE + (unsigned)r3 * SSZ + c3;
      float lmax = -1e30f;
      for (int s = 0; s < 256; s += 8) {
        const v8h v = *(const v8h*)(pr + s);
#pragma unroll
        for (int j = 0; j < 8; ++j) lmax = fmaxf(lmax, (float)v[j]);
      }
      red[tid] = lmax;
    }
    __syncthreads();
    if (tid < 16) {
      float m = red[tid * 8];
#pragma unroll
      for (int j = 1; j < 8; ++j) m = fmaxf(m, red[tid * 8 + j]);
      rowmax[tid] = m;
    }
    __syncthreads();
    {
      f16* pr = pE + (unsigned)r3 * SSZ + c3;
      const float rm = rowmax[r3];
      float lsum = 0.f;
      for (int s = 0; s < 256; s += 8) {
        const v8h v = *(const v8h*)(pr + s);
        v8h e;
#pragma unroll
        for (int j = 0; j < 8; ++j) {
          const float ex = __expf((float)v[j] - rm);
          e[j] = (f16)ex;
          lsum += ex;
        }
        *(v8h*)(pr + s) = e;
      }
      red[tid] = lsum;
    }
    __syncthreads();
    if (tid < 16) {
      float ssum = 0.f;
#pragma unroll
      for (int j = 0; j < 8; ++j) ssum += red[tid * 8 + j];
      rowinv[tid] = 1.0f / ssum;
    }
    __syncthreads();

    // ---- Phase 3a: head-average accumulated in LDS (vectorized, no global)
    {
      const f16* pr = pE + (unsigned)r3 * SSZ + c3;
      float* ar = avg + (unsigned)r3 * SSZ + c3;
      const float riH = rowinv[r3] * invH;
      if (h == 0) {
        for (int s = 0; s < 256; s += 8) {
          const v8h e = *(const v8h*)(pr + s);
          v4f a0, a1;
#pragma unroll
          for (int j = 0; j < 4; ++j) {
            a0[j] = (float)e[j] * riH;
            a1[j] = (float)e[4 + j] * riH;
          }
          *(v4f*)(ar + s) = a0;
          *(v4f*)(ar + s + 4) = a1;
        }
      } else {
        for (int s = 0; s < 256; s += 8) {
          const v8h e = *(const v8h*)(pr + s);
          v4f a0 = *(const v4f*)(ar + s);
          v4f a1 = *(const v4f*)(ar + s + 4);
#pragma unroll
          for (int j = 0; j < 4; ++j) {
            a0[j] += (float)e[j] * riH;
            a1[j] += (float)e[4 + j] * riH;
          }
          *(v4f*)(ar + s) = a0;
          *(v4f*)(ar + s + 4) = a1;
        }
      }
    }

    // ---- Phase 3b: O_tile(16x16/wave) = E(16xS) @ V(Sx16); normalize at store
    const int d0 = wave * 16;
    const unsigned voff =
        ((unsigned)(b * NH + h) * HD2 + d0 + ln) * (SSZ * 2);   // bytes
    const f16* prow = pE + (unsigned)ln * SSZ;
    v8f oc = {};
    for (int ss0 = 0; ss0 < SSZ; ss0 += 32) {
      const v16h ap = load_a_lds(prow + ss0, g);         // ds_load_b128 x2
      oc = wmma16(ap, load_b_g(Vt, voff + ss0 * 2, g), oc);
    }
#pragma unroll
    for (int r = 0; r < 8; ++r) {
      const int m = r + g * 8;
      const float o = oc[r] * rowinv[m];
      Oh[((unsigned)(t0 + m) * BB + b) * DIM + h * HD2 + d0 + ln] = (f16)o;
    }
    __syncthreads();   // pE reused by next head
  }

  // ---- Final: vectorized write of the head-averaged attention
  {
    const float* ar = avg + (unsigned)r3 * SSZ + c3;
    const unsigned o = ((unsigned)b * TT + t0 + r3) * SSZ + c3;
    for (int s = 0; s < 256; s += 4)
      *(v4f*)(attn_avg + o + s) = *(const v4f*)(ar + s);
  }
}

// ---------------------------------------------------------------------------
extern "C" void kernel_launch(void* const* d_in, const int* in_sizes, int n_in,
                              void* d_out, int out_size, void* d_ws, size_t ws_size,
                              hipStream_t stream) {
  const float* query = (const float*)d_in[0];
  const float* key   = (const float*)d_in[1];
  const float* value = (const float*)d_in[2];
  const float* Wq = (const float*)d_in[3];  const float* bq = (const float*)d_in[4];
  const float* Wk = (const float*)d_in[5];  const float* bk = (const float*)d_in[6];
  const float* Wv = (const float*)d_in[7];  const float* bv = (const float*)d_in[8];
  const float* Wo = (const float*)d_in[9];  const float* bo = (const float*)d_in[10];

  const size_t NA = (size_t)MROWS * DIM;    // 4M activation elems
  const size_t NW = (size_t)DIM * DIM;      // 1M weight elems

  f16* qf  = (f16*)d_ws;                    // f16 copies of f32 inputs
  f16* kf  = qf + NA;
  f16* vf  = kf + NA;
  f16* wqh = vf + NA;
  f16* wkh = wqh + NW;
  f16* wvh = wkh + NW;
  f16* woh = wvh + NW;
  f16* Qh  = woh + NW;                      // [T,B,D]    f16
  f16* Kh  = Qh + NA;                       // [S,B,D]    f16
  f16* Vt  = Kh + NA;                       // [B,H,HD,S] f16
  f16* Oh  = Vt + NA;                       // [T,B,D]    f16

  float* Z        = (float*)d_out;          // [T,B,D]  f32
  float* attn_avg = Z + NA;                 // [B,T,S]  f32

  // f32 -> f16 staging (one pass; everything then lives in L2 as f16)
  cvt_kernel<<<dim3(NA / 2048), 256, 0, stream>>>(query, qf, (int)NA);
  cvt_kernel<<<dim3(NA / 2048), 256, 0, stream>>>(key,   kf, (int)NA);
  cvt_kernel<<<dim3(NA / 2048), 256, 0, stream>>>(value, vf, (int)NA);
  cvt_kernel<<<dim3(NW / 2048), 256, 0, stream>>>(Wq, wqh, (int)NW);
  cvt_kernel<<<dim3(NW / 2048), 256, 0, stream>>>(Wk, wkh, (int)NW);
  cvt_kernel<<<dim3(NW / 2048), 256, 0, stream>>>(Wv, wvh, (int)NW);
  cvt_kernel<<<dim3(NW / 2048), 256, 0, stream>>>(Wo, woh, (int)NW);

  const dim3 gg(MROWS / 128, DIM / 64);
  const float scaling = 0.125f;             // 64^-0.5

  gemm_kernel<0><<<gg, 128, 0, stream>>>(qf, wqh, bq, Qh, scaling);
  gemm_kernel<1><<<gg, 128, 0, stream>>>(kf, wkh, bk, Kh, 1.0f);
  gemm_kernel<2><<<gg, 128, 0, stream>>>(vf, wvh, bv, Vt, 1.0f);

  // LDS: pE f16 64KB + avg f32 128KB + scratch  (~193 KB < 320 KB/WGP)
  const size_t smem = (size_t)(24 * SSZ + 128 + 32) * sizeof(float);
  attn_kernel<<<dim3(TT / 16, BB), 128, smem, stream>>>(Qh, Kh, Vt, Oh, attn_avg);

  gemm_kernel<3><<<gg, 128, 0, stream>>>(Oh, woh, bo, Z, 1.0f);
}
